// CrossWindowAttention_44444321579412
// MI455X (gfx1250) — matrix operand order repaired and checked by reading, not compiled, not run on hardware
//
#include <hip/hip_runtime.h>

// ---------------- types ----------------
typedef __attribute__((ext_vector_type(16))) __bf16          v16bf;
typedef __attribute__((ext_vector_type(8)))  float           v8f;
typedef __attribute__((ext_vector_type(8)))  unsigned short  u16x8;
typedef __attribute__((ext_vector_type(4)))  int             v4i;

union FragCvt { u16x8 h[2]; v16bf v; };

// gfx1250 async global->LDS path (guarded; falls back to manual copies)
#if defined(__has_builtin)
#if __has_builtin(__builtin_amdgcn_global_load_async_to_lds_b128) && \
    __has_builtin(__builtin_amdgcn_s_wait_asynccnt)
#define USE_ASYNC 1
#endif
#endif
#ifndef USE_ASYNC
#define USE_ASYNC 0
#endif

#if USE_ASYNC
typedef __attribute__((address_space(1))) v4i g_v4i;   // global int4
typedef __attribute__((address_space(3))) v4i l_v4i;   // LDS int4
__device__ __forceinline__ void async_b128(const void* gsrc, void* ldst) {
    __builtin_amdgcn_global_load_async_to_lds_b128(
        (g_v4i*)(gsrc), (l_v4i*)(ldst), 0, 0);
}
__device__ __forceinline__ void async_wait0() {
    __builtin_amdgcn_s_wait_asynccnt(0);
}
#endif

__device__ __forceinline__ unsigned short f2bf(float f) {
    unsigned int u = __builtin_bit_cast(unsigned int, f);
    u += 0x7FFFu + ((u >> 16) & 1u);          // round-to-nearest-even
    return (unsigned short)(u >> 16);
}

#if defined(__has_builtin) && __has_builtin(__builtin_amdgcn_cvt_pk_bf16_f32)
typedef __attribute__((ext_vector_type(2))) __bf16 v2bf;
__device__ __forceinline__ unsigned int pk2bf(float a, float b) {
    v2bf r = __builtin_amdgcn_cvt_pk_bf16_f32(a, b);
    return __builtin_bit_cast(unsigned int, r);
}
#else
__device__ __forceinline__ unsigned int pk2bf(float a, float b) {
    return (unsigned int)f2bf(a) | ((unsigned int)f2bf(b) << 16);
}
#endif

__device__ __forceinline__ v8f vzero8() {
    v8f z = {0.f, 0.f, 0.f, 0.f, 0.f, 0.f, 0.f, 0.f};
    return z;
}

// A-fragment (16x32 bf16): lane m holds K = [klo..klo+7] and [klo+16..klo+23]
__device__ __forceinline__ v16bf load_a_frag(const unsigned short* rowptr, int klo) {
    FragCvt f;
    f.h[0] = *(const u16x8*)(rowptr + klo);
    f.h[1] = *(const u16x8*)(rowptr + klo + 16);
    return f.v;
}

// B-fragment (32x16 bf16): lane n holds K = [k0..k0+7] and [k0+16..k0+23]
__device__ __forceinline__ v16bf load_b_frag(const unsigned short* rowptr, int k0) {
    FragCvt f;
    f.h[0] = *(const u16x8*)(rowptr + k0);
    f.h[1] = *(const u16x8*)(rowptr + k0 + 16);
    return f.v;
}

// ---------------- fp32 -> bf16 packed conversion kernel (weights, once) ------
__global__ __launch_bounds__(256)
void cvt_bf16(const float* __restrict__ src, unsigned short* __restrict__ dst) {
    const int i = (blockIdx.x * 256 + threadIdx.x) * 2;
    float2 f = *(const float2*)(src + i);
    *(unsigned int*)(dst + i) = pk2bf(f.x, f.y);
}

// ---------------- GEMM: Y[M,512] = (X[M,512] @ Wb[512,512]^T + bias) * scale -
// Wb is bf16. 128x128 block tile, K-chunks of 64, 8 waves (4M x 2N),
// wave tile 32x64. bf16 tiles staged via async global->LDS when available.
template <bool IN_BF16, bool OUT_BF16>
__global__ __launch_bounds__(256)
void gemm512(const void* __restrict__ Xv, const unsigned short* __restrict__ Wb,
             const float* __restrict__ bias, void* __restrict__ Yv,
             float scale, int M) {
    constexpr int KD = 512;
    __shared__ __align__(16) unsigned short As[128][64];
    __shared__ __align__(16) unsigned short Bs[128][64];

    const int tid  = threadIdx.x;
    const int lane = tid & 31;
    const int wid  = tid >> 5;
    const int wm   = (wid >> 1) * 32;   // wave M offset inside tile
    const int wn   = (wid & 1) * 64;    // wave N offset inside tile
    const int m0   = blockIdx.y * 128;
    const int n0   = blockIdx.x * 128;

    const float*          Xf = (const float*)Xv;
    const unsigned short* Xb = (const unsigned short*)Xv;

    v8f acc[2][4];
    for (int i = 0; i < 2; ++i)
        for (int j = 0; j < 4; ++j) acc[i][j] = vzero8();

    for (int kc = 0; kc < KD / 64; ++kc) {
        const int k0 = kc * 64;

        // stage W tile (128x64 bf16)
#if USE_ASYNC
        for (int i = 0; i < 4; ++i) {
            int idx = tid + i * 256;          // 1024 x b128
            int r = idx >> 3, c = idx & 7;
            async_b128(Wb + (size_t)(n0 + r) * KD + k0 + c * 8, &Bs[r][c * 8]);
        }
#else
        for (int i = 0; i < 4; ++i) {
            int idx = tid + i * 256;
            int r = idx >> 3, c = idx & 7;
            *(uint4*)(&Bs[r][c * 8]) =
                *(const uint4*)(Wb + (size_t)(n0 + r) * KD + k0 + c * 8);
        }
#endif
        // stage X tile (128x64) into LDS as bf16
        if (IN_BF16) {
#if USE_ASYNC
            for (int i = 0; i < 4; ++i) {
                int idx = tid + i * 256;
                int r = idx >> 3, c = idx & 7;
                async_b128(Xb + (size_t)(m0 + r) * KD + k0 + c * 8, &As[r][c * 8]);
            }
#else
            for (int i = 0; i < 4; ++i) {
                int idx = tid + i * 256;
                int r = idx >> 3, c = idx & 7;
                *(uint4*)(&As[r][c * 8]) =
                    *(const uint4*)(Xb + (size_t)(m0 + r) * KD + k0 + c * 8);
            }
#endif
        } else {
            for (int i = 0; i < 8; ++i) {
                int idx = tid + i * 256;          // 2048 float4
                int r = idx >> 4, c = idx & 15;
                float4 f = *(const float4*)(Xf + (size_t)(m0 + r) * KD + k0 + c * 4);
                uint2 p;
                p.x = pk2bf(f.x, f.y);
                p.y = pk2bf(f.z, f.w);
                *(uint2*)(&As[r][c * 4]) = p;
            }
        }
#if USE_ASYNC
        async_wait0();
#endif
        __syncthreads();

        // prefetch next K-chunk of the fp32 activations while MACs run
        if (!IN_BF16 && (kc + 1 < KD / 64)) {
            const int pr = tid >> 1;              // 0..127
            __builtin_prefetch(Xf + (size_t)(m0 + pr) * KD + k0 + 64, 0, 0);
        }

        for (int ks = 0; ks < 2; ++ks) {
            v16bf afr[2], bfr[4];
            const int klo = ks * 32 + ((lane < 16) ? 0 : 8);
            const int kb  = ks * 32 + ((lane < 16) ? 0 : 8);
            for (int ti = 0; ti < 2; ++ti)
                afr[ti] = load_a_frag(&As[wm + ti * 16 + (lane & 15)][0], klo);
            for (int tj = 0; tj < 4; ++tj)
                bfr[tj] = load_b_frag(&Bs[wn + tj * 16 + (lane & 15)][0], kb);
            for (int ti = 0; ti < 2; ++ti)
                for (int tj = 0; tj < 4; ++tj)
                    acc[ti][tj] = __builtin_amdgcn_wmma_f32_16x16x32_bf16(
                        false, afr[ti], false, bfr[tj], (short)0, acc[ti][tj],
                        false, false);
        }
        __syncthreads();
    }

    // epilogue: +bias, *scale, store
    for (int ti = 0; ti < 2; ++ti) {
        const int rbase = m0 + wm + ti * 16 + ((lane >> 4) << 3);
        for (int tj = 0; tj < 4; ++tj) {
            const int colg = n0 + wn + tj * 16 + (lane & 15);
            const float bv = bias[colg];
            for (int j = 0; j < 8; ++j) {
                float val = (acc[ti][tj][j] + bv) * scale;
                const size_t off = (size_t)(rbase + j) * KD + colg;
                if (OUT_BF16) ((unsigned short*)Yv)[off] = f2bf(val);
                else          ((float*)Yv)[off]          = val;
            }
        }
    }
}

// ---------------- fused window attention ----------------
// One block per (b, h). 4 waves. Q already scaled. O overwrites Q region.
__global__ __launch_bounds__(128)
void winattn(const unsigned short* __restrict__ Qb,
             const unsigned short* __restrict__ Kb,
             const unsigned short* __restrict__ Vb,
             const float* __restrict__ mask,   // [64,64,64]
             const float* __restrict__ btab,   // [225,16]
             unsigned short* __restrict__ Ob) {
    const int h    = blockIdx.x;      // 0..15
    const int b    = blockIdx.y;      // 0..1023
    const int tid  = threadIdx.x;
    const int lane = tid & 31;
    const int mt   = tid >> 5;        // wave id = M-tile 0..3

    __shared__ float                        Ss[64][65];   // scores fp32 (padded)
    __shared__ __align__(16) unsigned short Ps[64][72];   // probs bf16 (padded)
    __shared__ __align__(16) unsigned short VTs[32][72];  // V^T bf16 (padded)
    __shared__ __align__(16) unsigned short Ks[64][48];   // K tile bf16 (padded)

    const size_t base = ((size_t)b * 64) * 512 + (size_t)h * 32;
    const unsigned short* Qp = Qb + base;
    const unsigned short* Kp = Kb + base;
    const unsigned short* Vp = Vb + base;

    // stage K tile (64x32 bf16) into LDS (shared by all 4 waves)
#if USE_ASYNC
    for (int i = 0; i < 2; ++i) {
        int idx = tid + i * 128;            // 256 x b128
        int r = idx >> 2, c = idx & 3;
        async_b128(Kp + (size_t)r * 512 + c * 8, &Ks[r][c * 8]);
    }
#else
    for (int i = 0; i < 2; ++i) {
        int idx = tid + i * 128;
        int r = idx >> 2, c = idx & 3;
        *(uint4*)(&Ks[r][c * 8]) = *(const uint4*)(Kp + (size_t)r * 512 + c * 8);
    }
#endif

    // stage V transposed: VTs[d][m] = V[m][d]
    {
        const int r  = tid >> 1;            // token m 0..63
        const int c0 = (tid & 1) * 16;      // d half
        const unsigned short* vp = Vp + (size_t)r * 512 + c0;
        for (int c = 0; c < 16; ++c) VTs[c0 + c][r] = vp[c];
    }

    // Q fragment straight from global (each wave owns its 16 rows)
    v16bf aq;
    {
        const int row = mt * 16 + (lane & 15);
        const int klo = (lane < 16) ? 0 : 8;
        aq = load_a_frag(Qp + (size_t)row * 512, klo);
    }
#if USE_ASYNC
    async_wait0();
#endif
    __syncthreads();

    // S = Q K^T  (A = Q rows, B = K rows; both K-contiguous)
    v8f sacc[4];
    for (int nt = 0; nt < 4; ++nt) {
        const int kb = (lane < 16) ? 0 : 8;
        v16bf bk = load_b_frag(&Ks[nt * 16 + (lane & 15)][0], kb);
        sacc[nt] = __builtin_amdgcn_wmma_f32_16x16x32_bf16(
            false, aq, false, bk, (short)0, vzero8(), false, false);
    }

    // + relative position bias + mask, scatter to LDS
    const float* mrow0 = mask + (size_t)(b & 63) * 64 * 64;
    for (int nt = 0; nt < 4; ++nt) {
        const int col = nt * 16 + (lane & 15);
        const int ym = col >> 3, xm = col & 7;
        for (int j = 0; j < 8; ++j) {
            const int row = mt * 16 + ((lane >> 4) << 3) + j;
            const int yn = row >> 3, xn = row & 7;
            const int idx = (yn - ym + 7) * 15 + (xn - xm + 7);
            Ss[row][col] = sacc[nt][j] + btab[idx * 16 + h] + mrow0[row * 64 + col];
        }
    }
    __syncthreads();

    // row softmax (64 rows, one thread each)
    if (tid < 64) {
        float mx = -3.4e38f;
        for (int c = 0; c < 64; ++c) mx = fmaxf(mx, Ss[tid][c]);
        float sum = 0.f;
        for (int c = 0; c < 64; ++c) {
            float e = __expf(Ss[tid][c] - mx);
            sum += e;
            Ss[tid][c] = e;
        }
        const float inv = 1.f / sum;
        for (int c = 0; c < 32; ++c) {
            *(unsigned int*)(&Ps[tid][c * 2]) =
                pk2bf(Ss[tid][c * 2] * inv, Ss[tid][c * 2 + 1] * inv);
        }
    }
    __syncthreads();

    // O = P @ V  (K = 64 -> two WMMA steps), d-tiles nt = 0,1
    v8f oacc[2];
    oacc[0] = vzero8();
    oacc[1] = vzero8();
    for (int ks = 0; ks < 2; ++ks) {
        const int klo = ks * 32 + ((lane < 16) ? 0 : 8);
        const int kb  = ks * 32 + ((lane < 16) ? 0 : 8);
        v16bf ap = load_a_frag(&Ps[mt * 16 + (lane & 15)][0], klo);
        for (int nt = 0; nt < 2; ++nt) {
            v16bf bv = load_b_frag(&VTs[nt * 16 + (lane & 15)][0], kb);
            oacc[nt] = __builtin_amdgcn_wmma_f32_16x16x32_bf16(
                false, ap, false, bv, (short)0, oacc[nt], false, false);
        }
    }

    // store O (layout [b, n, h, hd] == [b*64+n, 512] with col h*32+d)
    for (int nt = 0; nt < 2; ++nt) {
        const int dcol = nt * 16 + (lane & 15);
        for (int j = 0; j < 8; ++j) {
            const int row = mt * 16 + ((lane >> 4) << 3) + j;
            Ob[base + (size_t)row * 512 + dcol] = f2bf(oacc[nt][j]);
        }
    }
}

// ---------------- launcher ----------------
extern "C" void kernel_launch(void* const* d_in, const int* in_sizes, int n_in,
                              void* d_out, int out_size, void* d_ws, size_t ws_size,
                              hipStream_t stream) {
    const float* q    = (const float*)d_in[0];
    const float* k    = (const float*)d_in[1];
    const float* v    = (const float*)d_in[2];
    const float* mask = (const float*)d_in[3];
    const float* Wq   = (const float*)d_in[4];
    const float* bq   = (const float*)d_in[5];
    const float* Wk   = (const float*)d_in[6];
    const float* bk   = (const float*)d_in[7];
    const float* Wv   = (const float*)d_in[8];
    const float* bv   = (const float*)d_in[9];
    const float* Wp   = (const float*)d_in[10];
    const float* bp   = (const float*)d_in[11];
    const float* btab = (const float*)d_in[12];

    const int M = in_sizes[0] / 512;                 // 65536 rows
    const float SCALE = 0.17677669529663687f;        // 32^-0.5
    const size_t WSZ = 512 * 512;                    // one weight matrix

    unsigned short* Wqb = (unsigned short*)d_ws;
    unsigned short* Wkb = Wqb + WSZ;
    unsigned short* Wvb = Wkb + WSZ;
    unsigned short* Wpb = Wvb + WSZ;
    unsigned short* Qb  = Wpb + WSZ;
    unsigned short* Kb  = Qb + (size_t)M * 512;
    unsigned short* Vb  = Kb + (size_t)M * 512;

    // one-time (per launch) weight conversion: 4 x 512x512 fp32 -> bf16
    const int cgrid = (int)(WSZ / 2 / 256);          // 512 blocks
    cvt_bf16<<<cgrid, 256, 0, stream>>>(Wq, Wqb);
    cvt_bf16<<<cgrid, 256, 0, stream>>>(Wk, Wkb);
    cvt_bf16<<<cgrid, 256, 0, stream>>>(Wv, Wvb);
    cvt_bf16<<<cgrid, 256, 0, stream>>>(Wp, Wpb);

    dim3 ggrid(512 / 128, M / 128);

    gemm512<false, true><<<ggrid, 256, 0, stream>>>(q, Wqb, bq, Qb, SCALE, M);
    gemm512<false, true><<<ggrid, 256, 0, stream>>>(k, Wkb, bk, Kb, 1.0f, M);
    gemm512<false, true><<<ggrid, 256, 0, stream>>>(v, Wvb, bv, Vb, 1.0f, M);

    winattn<<<dim3(16, M / 64), 128, 0, stream>>>(Qb, Kb, Vb, mask, btab, Qb);

    gemm512<true, false><<<ggrid, 256, 0, stream>>>(Qb, Wpb, bp, (float*)d_out, 1.0f, M);
}
